// Renderer_72275709657613
// MI455X (gfx1250) — matrix operand order repaired and checked by reading, not compiled, not run on hardware
//
#include <hip/hip_runtime.h>
#include <hip/hip_bf16.h>
#include <math.h>

// ---------------- problem constants (from reference setup_inputs) ----------
#define B_        4
#define R_        8192
#define NS_       48
#define RES_      48
#define C_        32
#define NEARV     0.0f
#define FARV      1.5f
#define AABB_MIN_ (-0.8f)
#define INV_AABB_ 1.25f
#define DT_       ((FARV - NEARV) / (float)NS_)   // 0.03125, uniform spacing

#define RAYS_PB   2
#define PTS_PB    (RAYS_PB * NS_)   // 96 points per block
#define THREADS   PTS_PB            // 96 threads = 3 wave32 waves
#define NTILES    (PTS_PB / 16)     // 6 row-tiles of 16 points
#define NWAVES    (THREADS / 32)    // 3
#define KPAD      96                // K padded to 3 x 32 for f16 WMMA

typedef _Float16 v16h __attribute__((ext_vector_type(16)));
typedef _Float16 v8h  __attribute__((ext_vector_type(8)));
typedef float    v8f  __attribute__((ext_vector_type(8)));

// Branch-free gelu on hardware v_exp_f32. Max abs error ~3e-4 -- at the same
// noise floor as the fp16 activation staging the WMMA path already imposes.
__device__ __forceinline__ float geluf(float x) {
  const float u  = 0.7978845608028654f * x * (1.0f + 0.044715f * x * x);
  const float e  = __expf(2.0f * u);
  const float th = 1.0f - 2.0f / (e + 1.0f);   // tanh(u), branch-free
  return 0.5f * x * (1.0f + th);
}

__device__ __forceinline__ float sigmoidf_fast(float x) {
  return 1.0f / (1.0f + __expf(-x));
}

// A-fragment (16x32 f16 tile) per ISA 7.12.2:
// lane&15 = row M; hi = lane>>4 selects K interleave half.
// element e: K = k0 + (e<8 ? hi*8+e : 16 + hi*8 + (e-8))
// -> two contiguous 16-byte runs per lane: [k0+hi*8, +8) and [k0+16+hi*8, +8)
__device__ __forceinline__ v16h load_a_frag(const _Float16* sA, int rowBase, int k0, int lane) {
  const int hi = lane >> 4;
  const _Float16* p = sA + (rowBase + (lane & 15)) * KPAD + k0 + hi * 8;
  const v8h lo = *(const v8h*)(p);
  const v8h hh = *(const v8h*)(p + 16);
  return __builtin_shufflevector(lo, hh, 0, 1, 2, 3, 4, 5, 6, 7, 8, 9, 10, 11, 12, 13, 14, 15);
}

// B-fragment (32x16 slice of KxN weight): column N = n0 + (lane&15); same K order
// as the A-fragment: [k0+hi*8, +8) and [k0+16+hi*8, +8). Weights staged TRANSPOSED
// in LDS as [N][kstride] so each run is contiguous -> two b128 loads.
__device__ __forceinline__ v16h load_b_fragT(const _Float16* sWt, int kstride, int k0, int n0, int lane) {
  const int col = n0 + (lane & 15);
  const int hi  = lane >> 4;
  const _Float16* p = sWt + col * kstride + k0 + hi * 8;
  const v8h lo = *(const v8h*)(p);
  const v8h hh = *(const v8h*)(p + 16);
  return __builtin_shufflevector(lo, hh, 0, 1, 2, 3, 4, 5, 6, 7, 8, 9, 10, 11, 12, 13, 14, 15);
}

__global__ __launch_bounds__(THREADS)
void renderer_fused_kernel(const float* __restrict__ rays_o, const float* __restrict__ rays_d,
                           const float* __restrict__ matrixs, const float* __restrict__ vectors,
                           const float* __restrict__ w_mat,   const float* __restrict__ b_mat,
                           const float* __restrict__ w1,      const float* __restrict__ b1,
                           const float* __restrict__ w2,      const float* __restrict__ b2,
                           float* __restrict__ out) {
  // Activation staging: 96 rows x 96 cols f16, reused by all 3 MLP layers.
  __shared__ __align__(16) _Float16 sA[PTS_PB * KPAD];   // 18 KB
  __shared__ __align__(16) _Float16 sW0t[64 * KPAD];     // 12 KB (w_mat^T, zero-pad K)
  __shared__ __align__(16) _Float16 sW1t[64 * KPAD];     // 12 KB (w1^T,    zero-pad K)
  __shared__ __align__(16) _Float16 sW2t[16 * 64];       // 2 KB  (w2^T, cols>=3 zero)
  __shared__ __align__(16) _Float16 sPE[RAYS_PB * 32];   // per-ray PE, padded to 32
  __shared__ float sBm[64], sB1v[64], sB2v[3];
  __shared__ float sSig[PTS_PB];
  __shared__ float sRgb[PTS_PB * 3];

  const int tid  = threadIdx.x;
  const int ray0 = blockIdx.x * RAYS_PB;

  // ---- stage weights/biases into LDS, transposed (cooperative) ----
  for (int idx = tid; idx < 64 * KPAD; idx += THREADS) {
    const int n = idx / KPAD, k = idx - n * KPAD;
    sW0t[idx] = (k < 72) ? (_Float16)w_mat[k * 64 + n] : (_Float16)0.0f;
    sW1t[idx] = (k < 91) ? (_Float16)w1[k * 64 + n]    : (_Float16)0.0f;
  }
  for (int idx = tid; idx < 16 * 64; idx += THREADS) {
    const int n = idx >> 6, k = idx & 63;
    sW2t[idx] = (n < 3) ? (_Float16)w2[k * 3 + n] : (_Float16)0.0f;
  }
  if (tid < 64) { sBm[tid] = b_mat[tid]; sB1v[tid] = b1[tid]; }
  if (tid < 3)  sB2v[tid] = b2[tid];

  // ---- featurize: one sample point per thread ----
  {
    const int lp  = tid;            // local point 0..95
    const int lr  = lp / NS_;       // local ray 0..1
    const int s   = lp % NS_;       // sample index
    const int ray = ray0 + lr;
    const int b   = ray / R_;

    const float ox = rays_o[ray * 3 + 0], oy = rays_o[ray * 3 + 1], oz = rays_o[ray * 3 + 2];
    const float dx = rays_d[ray * 3 + 0], dy = rays_d[ray * 3 + 1], dz = rays_d[ray * 3 + 2];
    const float t  = NEARV + ((float)s + 0.5f) * DT_;

    float xyz[3];
    xyz[0] = ((ox + dx * t) - AABB_MIN_) * INV_AABB_ - 1.0f;
    xyz[1] = ((oy + dy * t) - AABB_MIN_) * INV_AABB_ - 1.0f;
    xyz[2] = ((oz + dz * t) - AABB_MIN_) * INV_AABB_ - 1.0f;

    // MAT_MODE = [[0,1],[2,0],[1,2]] (x-coord, y-coord); VEC_MODE = [2,1,0]
    const int m0c[3] = {0, 2, 1};
    const int m1c[3] = {1, 0, 2};
    const int vcm[3] = {2, 1, 0};

    float sig = 0.0f;
#pragma unroll
    for (int i = 0; i < 3; ++i) {
      // --- 2D bilinear sample setup ---
      const float cx = xyz[m0c[i]], cy = xyz[m1c[i]];
      const float fx = (cx + 1.0f) * 0.5f * (float)(RES_ - 1);
      const float fy = (cy + 1.0f) * 0.5f * (float)(RES_ - 1);
      const float x0f = floorf(fx), y0f = floorf(fy);
      const float wx = fx - x0f, wy = fy - y0f;
      const int x0 = (int)x0f, y0 = (int)y0f, x1 = x0 + 1, y1 = y0 + 1;
      const float mx0 = (x0 >= 0 && x0 < RES_) ? 1.0f : 0.0f;
      const float mx1 = (x1 >= 0 && x1 < RES_) ? 1.0f : 0.0f;
      const float my0 = (y0 >= 0 && y0 < RES_) ? 1.0f : 0.0f;
      const float my1 = (y1 >= 0 && y1 < RES_) ? 1.0f : 0.0f;
      const int cx0 = min(max(x0, 0), RES_ - 1), cx1 = min(max(x1, 0), RES_ - 1);
      const int cy0 = min(max(y0, 0), RES_ - 1), cy1 = min(max(y1, 0), RES_ - 1);
      const float w00 = (1.0f - wx) * (1.0f - wy) * mx0 * my0;
      const float w10 = wx * (1.0f - wy) * mx1 * my0;
      const float w01 = (1.0f - wx) * wy * mx0 * my1;
      const float w11 = wx * wy * mx1 * my1;
      const int o00 = cy0 * RES_ + cx0, o10 = cy0 * RES_ + cx1;
      const int o01 = cy1 * RES_ + cx0, o11 = cy1 * RES_ + cx1;

      // --- 1D linear sample setup ---
      const float cz = xyz[vcm[i]];
      const float fz = (cz + 1.0f) * 0.5f * (float)(RES_ - 1);
      const float z0f = floorf(fz);
      const float wz = fz - z0f;
      const int z0 = (int)z0f, z1 = z0 + 1;
      const float v0 = (1.0f - wz) * ((z0 >= 0 && z0 < RES_) ? 1.0f : 0.0f);
      const float v1 = wz * ((z1 >= 0 && z1 < RES_) ? 1.0f : 0.0f);
      const int cz0 = min(max(z0, 0), RES_ - 1), cz1 = min(max(z1, 0), RES_ - 1);

      const float* p2 = matrixs + (size_t)((i * B_ + b) * C_) * (RES_ * RES_);
      const float* p1 = vectors + (size_t)((i * B_ + b) * C_) * RES_;

      // sigma channels 0..7
#pragma unroll
      for (int c = 0; c < 8; ++c) {
        const float* fc = p2 + c * (RES_ * RES_);
        const float pf = w00 * fc[o00] + w10 * fc[o10] + w01 * fc[o01] + w11 * fc[o11];
        const float* vp = p1 + c * RES_;
        const float lf = v0 * vp[cz0] + v1 * vp[cz1];
        sig += pf * lf;
      }
      // rgb feature channels 8..31, staged as inner-gelu f16, 8 at a time (b128 stores)
#pragma unroll
      for (int cg = 0; cg < 3; ++cg) {
        v8h chunk;
#pragma unroll
        for (int j = 0; j < 8; ++j) {
          const int c = 8 + cg * 8 + j;
          const float* fc = p2 + c * (RES_ * RES_);
          const float pf = w00 * fc[o00] + w10 * fc[o10] + w01 * fc[o01] + w11 * fc[o11];
          const float* vp = p1 + c * RES_;
          const float lf = v0 * vp[cz0] + v1 * vp[cz1];
          chunk[j] = (_Float16)geluf(pf * lf);
        }
        *(v8h*)(sA + lp * KPAD + i * 24 + cg * 8) = chunk;
      }
    }
    sSig[lp] = fmaxf(sig, 0.0f);
    // zero-pad cols 72..95 for layer-1 K padding (3 x b128)
    {
      const v8h z = {};
      *(v8h*)(sA + lp * KPAD + 72) = z;
      *(v8h*)(sA + lp * KPAD + 80) = z;
      *(v8h*)(sA + lp * KPAD + 88) = z;
    }

    // per-ray positional encoding of normalized view dir: [d(3), sin(12), cos(12), pad(5)]
    if (s == 0) {
      const float inv = rsqrtf(dx * dx + dy * dy + dz * dz);
      const float v3[3] = {dx * inv, dy * inv, dz * inv};
      _Float16* pe = sPE + lr * 32;
      pe[0] = (_Float16)v3[0]; pe[1] = (_Float16)v3[1]; pe[2] = (_Float16)v3[2];
#pragma unroll
      for (int f = 0; f < 4; ++f) {
        const float fr = (float)(1 << f);
#pragma unroll
        for (int d = 0; d < 3; ++d) {
          const float ang = v3[d] * fr;
          pe[3 + f * 3 + d]  = (_Float16)__sinf(ang);
          pe[15 + f * 3 + d] = (_Float16)__cosf(ang);
        }
      }
#pragma unroll
      for (int j = 27; j < 32; ++j) pe[j] = (_Float16)0.0f;
    }
  }
  __syncthreads();

  // ---- MLP: each wave owns tiles {wave, wave+NWAVES}; rows are wave-exclusive,
  //      so inter-layer LDS reuse relies only on per-wave DS ordering. ----
  {
    const int lane = tid & 31;
    const int wave = tid >> 5;
    const int hi   = lane >> 4;

    for (int tile = wave; tile < NTILES; tile += NWAVES) {
      const int base = tile * 16;

      // ----- layer 1: [16x96] x [96x64] -----
      v16h a0 = load_a_frag(sA, base, 0,  lane);
      v16h a1 = load_a_frag(sA, base, 32, lane);
      v16h a2 = load_a_frag(sA, base, 64, lane);
#pragma unroll
      for (int nt = 0; nt < 4; ++nt) {
        const int n0 = nt * 16;
        v8f acc = {};
        acc = __builtin_amdgcn_wmma_f32_16x16x32_f16(false, a0, false, load_b_fragT(sW0t, KPAD, 0,  n0, lane), (short)0, acc, false, false);
        acc = __builtin_amdgcn_wmma_f32_16x16x32_f16(false, a1, false, load_b_fragT(sW0t, KPAD, 32, n0, lane), (short)0, acc, false, false);
        acc = __builtin_amdgcn_wmma_f32_16x16x32_f16(false, a2, false, load_b_fragT(sW0t, KPAD, 64, n0, lane), (short)0, acc, false, false);
        const int col = n0 + (lane & 15);
        const float bias = sBm[col];
#pragma unroll
        for (int rr = 0; rr < 8; ++rr) {
          sA[(base + rr + hi * 8) * KPAD + col] = (_Float16)geluf(acc[rr] + bias);
        }
      }

      // splice view-dir PE (constant within tile: 48 % 16 == 0) into cols 64..95
      {
        const int lr = base / NS_;
        const v8h* pp = (const v8h*)(sPE + lr * 32 + hi * 16);
        const v8h pe0 = pp[0];
        const v8h pe1 = pp[1];
        v8h* dst = (v8h*)(sA + (base + (lane & 15)) * KPAD + 64 + hi * 16);
        dst[0] = pe0;
        dst[1] = pe1;
      }

      // ----- layer 2: [16x96] x [96x64] -----
      v16h h0 = load_a_frag(sA, base, 0,  lane);
      v16h h1 = load_a_frag(sA, base, 32, lane);
      v16h h2 = load_a_frag(sA, base, 64, lane);
#pragma unroll
      for (int nt = 0; nt < 4; ++nt) {
        const int n0 = nt * 16;
        v8f acc = {};
        acc = __builtin_amdgcn_wmma_f32_16x16x32_f16(false, h0, false, load_b_fragT(sW1t, KPAD, 0,  n0, lane), (short)0, acc, false, false);
        acc = __builtin_amdgcn_wmma_f32_16x16x32_f16(false, h1, false, load_b_fragT(sW1t, KPAD, 32, n0, lane), (short)0, acc, false, false);
        acc = __builtin_amdgcn_wmma_f32_16x16x32_f16(false, h2, false, load_b_fragT(sW1t, KPAD, 64, n0, lane), (short)0, acc, false, false);
        const int col = n0 + (lane & 15);
        const float bias = sB1v[col];
#pragma unroll
        for (int rr = 0; rr < 8; ++rr) {
          sA[(base + rr + hi * 8) * KPAD + col] = (_Float16)geluf(acc[rr] + bias);
        }
      }

      // ----- layer 3: [16x64] x [64x16] (w2 cols >= 3 are zero) -----
      v16h g0 = load_a_frag(sA, base, 0,  lane);
      v16h g1 = load_a_frag(sA, base, 32, lane);
      v8f acc = {};
      acc = __builtin_amdgcn_wmma_f32_16x16x32_f16(false, g0, false, load_b_fragT(sW2t, 64, 0,  0, lane), (short)0, acc, false, false);
      acc = __builtin_amdgcn_wmma_f32_16x16x32_f16(false, g1, false, load_b_fragT(sW2t, 64, 32, 0, lane), (short)0, acc, false, false);
      const int col = lane & 15;
      if (col < 3) {
        const float bias = sB2v[col];
#pragma unroll
        for (int rr = 0; rr < 8; ++rr) {
          sRgb[(base + rr + hi * 8) * 3 + col] = sigmoidf_fast(acc[rr] + bias);
        }
      }
    }
  }
  __syncthreads();

  // ---- per-ray alpha compositing ----
  if (tid < RAYS_PB) {
    const int ray = ray0 + tid;
    float* outRgb   = out;
    float* outDepth = out + (size_t)B_ * R_ * 3;
    float* outW     = out + (size_t)B_ * R_ * 3 + (size_t)B_ * R_;

    float T = 1.0f, r0 = 0.0f, r1 = 0.0f, r2 = 0.0f, dm = 0.0f;
    for (int s = 0; s < NS_; ++s) {
      const float sg    = sSig[tid * NS_ + s];
      const float alpha = 1.0f - __expf(-sg * DT_);
      const float w     = alpha * T;
      outW[(size_t)ray * NS_ + s] = w;
      T *= (1.0f - alpha + 1e-10f);
      const float mid = NEARV + ((float)s + 0.5f) * DT_;
      dm += w * mid;
      const int p = tid * NS_ + s;
      r0 += w * sRgb[p * 3 + 0];
      r1 += w * sRgb[p * 3 + 1];
      r2 += w * sRgb[p * 3 + 2];
    }
    outRgb[(size_t)ray * 3 + 0] = r0;
    outRgb[(size_t)ray * 3 + 1] = r1;
    outRgb[(size_t)ray * 3 + 2] = r2;
    outDepth[ray] = dm;
  }
}

extern "C" void kernel_launch(void* const* d_in, const int* in_sizes, int n_in,
                              void* d_out, int out_size, void* d_ws, size_t ws_size,
                              hipStream_t stream) {
  (void)in_sizes; (void)n_in; (void)d_ws; (void)ws_size; (void)out_size;
  const float* rays_o  = (const float*)d_in[0];
  const float* rays_d  = (const float*)d_in[1];
  const float* matrixs = (const float*)d_in[2];
  const float* vectors = (const float*)d_in[3];
  const float* w_mat   = (const float*)d_in[4];
  const float* b_mat   = (const float*)d_in[5];
  const float* w1      = (const float*)d_in[6];
  const float* b1      = (const float*)d_in[7];
  const float* w2      = (const float*)d_in[8];
  const float* b2      = (const float*)d_in[9];
  float* out = (float*)d_out;

  const int nblocks = (B_ * R_) / RAYS_PB;  // 16384 blocks of 96 threads (3 waves)
  renderer_fused_kernel<<<dim3(nblocks), dim3(THREADS), 0, stream>>>(
      rays_o, rays_d, matrixs, vectors, w_mat, b_mat, w1, b1, w2, b2, out);
}
